// Renderer_fast2_90065464197139
// MI455X (gfx1250) — matrix operand order repaired and checked by reading, not compiled, not run on hardware
//
#include <hip/hip_runtime.h>
#include <hip/hip_bf16.h>

// ---------------------------------------------------------------------------
// Point-based neural renderer for MI455X (gfx1250, wave32, WMMA).
// f16 activations + f32 accumulation via v_wmma_f32_16x16x32_f16.
// Conv halo staging uses async global->LDS (ASYNCcnt) on CDNA5.
// ---------------------------------------------------------------------------

static constexpr int kH = 512, kW = 512, kP = 8, kNPTS = 100000;
static constexpr int kTOTAL = kH * kW * kP;          // 2,097,152

typedef __attribute__((ext_vector_type(16))) _Float16 v16h;
typedef __attribute__((ext_vector_type(8)))  float    v8f;

union Frag {
    v16h     h;
    unsigned u[8];
};

// A-matrix (16x32 f16) lane/VGPR -> K mapping (ISA 7.12.2):
// lanes 0-15: V0..3 hold K 0..7, V4..7 hold K 16..23 ; lanes 16-31: +8
__device__ __forceinline__ int kmapA(int v, int half) {
    return ((v < 4) ? (2 * v) : (16 + 2 * (v - 4))) + 8 * half;
}
// B-matrix (32x16 f16): same K mapping as A (lanes 0-15: V0..3 K 0..7, V4..7 K 16..23; lanes 16-31: +8)
__device__ __forceinline__ int kmapB(int v, int half) {
    return ((v < 4) ? (2 * v) : (16 + 2 * (v - 4))) + 8 * half;
}

__device__ __forceinline__ unsigned pack2(float a, float b) {
    unsigned short ua = __builtin_bit_cast(unsigned short, (_Float16)a);
    unsigned short ub = __builtin_bit_cast(unsigned short, (_Float16)b);
    return (unsigned)ua | ((unsigned)ub << 16);
}

__device__ __forceinline__ v8f wmma_f16(const Frag& a, const Frag& b, v8f c) {
    return __builtin_amdgcn_wmma_f32_16x16x32_f16(
        /*neg_a=*/false, a.h, /*neg_b=*/false, b.h,
        /*c_mod=*/(short)0, c, /*reuse_a=*/false, /*reuse_b=*/false);
}

// ---------------------------------------------------------------------------
// Phase 1: segment_min of first-visible index per point id
// ---------------------------------------------------------------------------
__global__ __launch_bounds__(256) void init_first(int* first) {
    int i = blockIdx.x * 256 + threadIdx.x;
    if (i < kNPTS) first[i] = 0x7fffffff;
}

__global__ __launch_bounds__(256) void seg_min(const float* __restrict__ z,
                                               const int* __restrict__ ids,
                                               const int* __restrict__ isTrain,
                                               int* __restrict__ first) {
    int i = blockIdx.x * 256 + threadIdx.x;           // i < kTOTAL (exact grid)
    float th = isTrain[0] ? 0.2f : 0.0f;
    bool  m  = z[i] > th;
    int   s  = m ? ids[i] : 0;
    int   vv = m ? i : kTOTAL;
    atomicMin(&first[s], vv);
}

// ---------------------------------------------------------------------------
// Weight repack into WMMA fragment layouts (f32 -> f16 pairs)
// ---------------------------------------------------------------------------
__global__ __launch_bounds__(256) void pack_conv_w(const float* __restrict__ w,
                                                   unsigned* __restrict__ dst,
                                                   int CIN, int COUTpad, int COUTreal) {
    int NT = COUTpad >> 4, KC = CIN >> 5;
    int total = 9 * NT * KC * 32 * 8;
    int idx = blockIdx.x * 256 + threadIdx.x;
    if (idx >= total) return;
    int v    = idx & 7;
    int lane = (idx >> 3) & 31;
    int rest = idx >> 8;
    int cc = rest % KC;  rest /= KC;
    int nt = rest % NT;  rest /= NT;
    int kk = rest;                                    // (ky*3+kx)
    int half = lane >> 4, n = lane & 15;
    int k  = kmapB(v, half);
    int ci = cc * 32 + k;
    int co = nt * 16 + n;
    float f0 = 0.f, f1 = 0.f;
    if (co < COUTreal) {
        f0 = w[((size_t)kk * CIN + ci)     * COUTreal + co];
        f1 = w[((size_t)kk * CIN + ci + 1) * COUTreal + co];
    }
    dst[idx] = pack2(f0, f1);
}

__global__ __launch_bounds__(256) void pack_mlp_w1(const float* __restrict__ w,
                                                   unsigned* __restrict__ dst) {
    int idx = blockIdx.x * 256 + threadIdx.x;         // 16*32*8 = 4096
    if (idx >= 16 * 32 * 8) return;
    int v = idx & 7, lane = (idx >> 3) & 31, nt = idx >> 8;
    int half = lane >> 4, n = lane & 15;
    int k = kmapB(v, half);
    float f0 = (k     < 6) ? w[(size_t)k       * 256 + nt * 16 + n] : 0.f;
    float f1 = (k + 1 < 6) ? w[(size_t)(k + 1) * 256 + nt * 16 + n] : 0.f;
    dst[idx] = pack2(f0, f1);
}

__global__ __launch_bounds__(256) void pack_mlp_w2(const float* __restrict__ w,
                                                   unsigned* __restrict__ dst) {
    int idx = blockIdx.x * 256 + threadIdx.x;         // 8*32*8 = 2048
    if (idx >= 8 * 32 * 8) return;
    int v = idx & 7, lane = (idx >> 3) & 31, kc = idx >> 8;
    int half = lane >> 4, n = lane & 15;
    int k = kc * 32 + kmapB(v, half);
    dst[idx] = pack2(w[(size_t)k * 16 + n], w[(size_t)(k + 1) * 16 + n]);
}

// ---------------------------------------------------------------------------
// Phase 2: per-point MLP 6->256->16 via WMMA (each wave owns 16 points)
// ---------------------------------------------------------------------------
__global__ __launch_bounds__(128)
void mlp_wmma(const float* __restrict__ zbufs, const float* __restrict__ ray,
              const int* __restrict__ first,
              const unsigned* __restrict__ w1p, const unsigned* __restrict__ w2p,
              const float* __restrict__ b1, const float* __restrict__ b2,
              float* __restrict__ feat) {
    __shared__ unsigned ldsA[4 * 16 * 16];    // per-wave 16x32 f16 A tile
    __shared__ unsigned ldsH[4 * 16 * 128];   // per-wave 16x256 f16 hidden tile
    int wave = threadIdx.x >> 5, lane = threadIdx.x & 31;
    int base = blockIdx.x * 64 + wave * 16;

    if (lane < 16) {
        float vals[6] = {0.f, 0.f, 0.f, 0.f, 0.f, 0.f};
        int p = base + lane;
        if (p < kNPTS) {
            int fi = first[p];
            fi = fi < 0 ? 0 : (fi > kTOTAL - 1 ? kTOTAL - 1 : fi);
            float z   = zbufs[fi];
            int   pix = fi >> 3;                      // / P
            const float* rp = ray + (size_t)pix * 7;
            float dx = rp[3], dy = rp[4], dz = rp[5], cs = rp[6];
            float t = z / cs;
            vals[0] = ray[0] + t * dx;
            vals[1] = ray[1] + t * dy;
            vals[2] = ray[2] + t * dz;
            vals[3] = dx; vals[4] = dy; vals[5] = dz;
        }
        int row = (wave * 16 + lane) * 16;
        ldsA[row + 0] = pack2(vals[0], vals[1]);
        ldsA[row + 1] = pack2(vals[2], vals[3]);
        ldsA[row + 2] = pack2(vals[4], vals[5]);
#pragma unroll
        for (int i = 3; i < 16; ++i) ldsA[row + i] = 0u;
    }
    __syncthreads();

    int half = lane >> 4, mr = lane & 15;
    Frag a;
#pragma unroll
    for (int v = 0; v < 8; ++v)
        a.u[v] = ldsA[(wave * 16 + mr) * 16 + (kmapA(v, half) >> 1)];

    v8f vzero = {};
    _Float16* hp = reinterpret_cast<_Float16*>(ldsH);
    for (int nt = 0; nt < 16; ++nt) {                 // layer 1: N = 256
        Frag b;
        const unsigned* wp = w1p + ((size_t)nt * 32 + lane) * 8;
#pragma unroll
        for (int v = 0; v < 8; ++v) b.u[v] = wp[v];
        v8f d = wmma_f16(a, b, vzero);
        float bb = b1[nt * 16 + mr];
#pragma unroll
        for (int r = 0; r < 8; ++r) {
            int m = r + 8 * half;
            float hv = fmaxf(d[r] + bb, 0.f);
            hp[(wave * 16 + m) * 256 + nt * 16 + mr] = (_Float16)hv;
        }
    }
    __syncthreads();

    v8f acc = vzero;                                  // layer 2: K = 256, N = 16
    for (int kc = 0; kc < 8; ++kc) {
        Frag ha;
#pragma unroll
        for (int v = 0; v < 8; ++v)
            ha.u[v] = ldsH[(wave * 16 + mr) * 128 + ((kc * 32 + kmapA(v, half)) >> 1)];
        Frag b;
        const unsigned* wp = w2p + ((size_t)kc * 32 + lane) * 8;
#pragma unroll
        for (int v = 0; v < 8; ++v) b.u[v] = wp[v];
        acc = wmma_f16(ha, b, acc);
    }
    float bb2 = b2[mr];
#pragma unroll
    for (int r = 0; r < 8; ++r) {
        int m = r + 8 * half;
        int p = base + m;
        if (p < kNPTS) feat[(size_t)p * 16 + mr] = acc[r] + bb2;
    }
}

// ---------------------------------------------------------------------------
// Phase 3: masked gather feat[ids] -> f16 feature image (512x512x128)
// ---------------------------------------------------------------------------
__global__ __launch_bounds__(256)
void gather_feat(const float* __restrict__ z, const int* __restrict__ ids,
                 const int* __restrict__ isTrain,
                 const float* __restrict__ feat, uint4* __restrict__ x16) {
    int e = blockIdx.x * 256 + threadIdx.x;           // e < kTOTAL (exact grid)
    float th = isTrain[0] ? 0.2f : 0.0f;
    bool  m  = z[e] > th;
    uint4 o0 = {0, 0, 0, 0}, o1 = {0, 0, 0, 0};
    if (m) {
        const float4* f = reinterpret_cast<const float4*>(feat + (size_t)ids[e] * 16);
        float4 f0 = f[0], f1 = f[1], f2 = f[2], f3 = f[3];
        o0.x = pack2(f0.x, f0.y); o0.y = pack2(f0.z, f0.w);
        o0.z = pack2(f1.x, f1.y); o0.w = pack2(f1.z, f1.w);
        o1.x = pack2(f2.x, f2.y); o1.y = pack2(f2.z, f2.w);
        o1.z = pack2(f3.x, f3.y); o1.w = pack2(f3.z, f3.w);
    }
    x16[(size_t)e * 2]     = o0;
    x16[(size_t)e * 2 + 1] = o1;
}

// ---------------------------------------------------------------------------
// Phase 4: 3x3 implicit-GEMM conv via WMMA.
// Block: 8x8 output pixels x COUT channels; 8 waves; LDS-staged 10x10xCIN halo
// copied with CDNA5 async global->LDS (ASYNCcnt), zero-filled at borders.
// EPI: 0 = relu->f16, 1 = x*sigmoid->f16 (gate), 2 = bias-only f32 (ch<3).
// ---------------------------------------------------------------------------
template <int CIN, int COUT, int EPI>
__global__ __launch_bounds__(256)
void conv3(const _Float16* __restrict__ src, const unsigned* __restrict__ wpack,
           const float* __restrict__ bias, _Float16* __restrict__ dst16,
           float* __restrict__ dst32, const _Float16* __restrict__ xsrc) {
    constexpr int NT     = COUT / 16;
    constexpr int KC     = CIN / 32;
    constexpr int NTILES = 4 * NT;                    // 4 M-tiles x NT N-tiles
    constexpr int TPW    = (NTILES >= 8) ? (NTILES / 8) : 1;
    __shared__ unsigned lds[(100 * CIN) / 2];         // 10x10xCIN f16 halo

    int tid = threadIdx.x;
    int by  = (blockIdx.x >> 6) << 3;                 // 64 patches per row
    int bx  = (blockIdx.x & 63) << 3;

    __builtin_prefetch(wpack, 0, 1);                  // global_prefetch_b8 (L2 warm)

    // Async-copy the halo straight into LDS: 16B per lane per issue, tracked by
    // ASYNCcnt (global_load_async_to_lds_b128). Border cells get ds-stored zeros.
    {
        constexpr int CHUNKS = (100 * CIN) / 8;       // 16-byte chunks
        unsigned ldsbase = (unsigned)(size_t)(&lds[0]);
        for (int c = tid; c < CHUNKS; c += 256) {
            int h0   = c * 8;                          // first f16 index of chunk
            int cell = h0 / CIN;
            int ch   = h0 - cell * CIN;
            int iy = cell / 10, ix = cell - iy * 10;
            int gy = by + iy - 1, gx = bx + ix - 1;
            unsigned laddr = ldsbase + (unsigned)(c * 16);
            if ((unsigned)gy < 512u && (unsigned)gx < 512u) {
                const _Float16* gp = src + ((size_t)(gy * 512 + gx)) * CIN + ch;
                unsigned long long ga = (unsigned long long)(size_t)gp;
                asm volatile("global_load_async_to_lds_b128 %0, %1, off"
                             :: "v"(laddr), "v"(ga) : "memory");
            } else {
                uint4 z4 = {0u, 0u, 0u, 0u};
                *reinterpret_cast<uint4*>(&lds[c * 4]) = z4;
            }
        }
        asm volatile("s_wait_asynccnt 0x0" ::: "memory");
    }
    __syncthreads();

    int wave = tid >> 5, lane = tid & 31;
    int half = lane >> 4, mr = lane & 15;
    bool active = (wave * TPW) < NTILES;
    int  nt     = active ? ((wave * TPW) >> 2) : 0;   // all of a wave's tiles share nt

    v8f vzero = {};
    v8f acc[TPW];
#pragma unroll
    for (int t = 0; t < TPW; ++t) acc[t] = vzero;

    if (active) {
        for (int kk = 0; kk < 9; ++kk) {
            int ky = kk / 3, kx = kk - ky * 3;
#pragma unroll
            for (int cc = 0; cc < KC; ++cc) {
                Frag b;
                const unsigned* wp =
                    wpack + ((((size_t)kk * NT + nt) * KC + cc) * 32 + lane) * 8;
#pragma unroll
                for (int v = 0; v < 8; ++v) b.u[v] = wp[v];
#pragma unroll
                for (int ti = 0; ti < TPW; ++ti) {
                    int T  = wave * TPW + ti;
                    int mt = T & 3;
                    int p  = mt * 16 + mr;            // output pixel within patch
                    int cell = ((p >> 3) + ky) * 10 + ((p & 7) + kx);
                    Frag a;
#pragma unroll
                    for (int v = 0; v < 8; ++v)
                        a.u[v] = lds[(cell * CIN + cc * 32 + kmapA(v, half)) >> 1];
                    acc[ti] = wmma_f16(a, b, acc[ti]);
                }
            }
        }

        int ch = nt * 16 + mr;
        float bv = 0.f;
        if constexpr (EPI == 2) { if (ch < 3) bv = bias[ch]; }
        else bv = bias[ch];

#pragma unroll
        for (int ti = 0; ti < TPW; ++ti) {
            int T  = wave * TPW + ti;
            int mt = T & 3;
#pragma unroll
            for (int r = 0; r < 8; ++r) {
                int m = r + 8 * half;
                int p = mt * 16 + m;
                int gy = by + (p >> 3), gx = bx + (p & 7);
                size_t pix = (size_t)(gy * 512 + gx);
                float val = acc[ti][r] + bv;
                if constexpr (EPI == 0) {
                    dst16[pix * COUT + ch] = (_Float16)fmaxf(val, 0.f);
                } else if constexpr (EPI == 1) {
                    float xv = (float)xsrc[pix * 128 + ch];
                    float s  = 1.f / (1.f + __expf(-val));
                    dst16[pix * COUT + ch] = (_Float16)(xv * s);
                } else {
                    if (ch < 3) dst32[pix * 3 + ch] = val;
                }
            }
        }
    }
}

// ---------------------------------------------------------------------------
// Host-side orchestration
// ---------------------------------------------------------------------------
extern "C" void kernel_launch(void* const* d_in, const int* in_sizes, int n_in,
                              void* d_out, int out_size, void* d_ws, size_t ws_size,
                              hipStream_t stream) {
    const float* zbufs   = (const float*)d_in[0];
    const int*   idbufs  = (const int*)d_in[1];
    const float* ray     = (const float*)d_in[2];
    const int*   isTrain = (const int*)d_in[5];
    const float* mlp_w1  = (const float*)d_in[7];
    const float* mlp_b1  = (const float*)d_in[8];
    const float* mlp_w2  = (const float*)d_in[9];
    const float* mlp_b2  = (const float*)d_in[10];
    const float* mpn_w1  = (const float*)d_in[11];
    const float* mpn_b1  = (const float*)d_in[12];
    const float* mpn_w2  = (const float*)d_in[13];
    const float* mpn_b2  = (const float*)d_in[14];
    const float* unet_w1 = (const float*)d_in[15];
    const float* unet_b1 = (const float*)d_in[16];
    const float* unet_w2 = (const float*)d_in[17];
    const float* unet_b2 = (const float*)d_in[18];
    float* img = (float*)d_out;

    char*  ws  = (char*)d_ws;
    size_t off = 0;
    auto alloc = [&](size_t bytes) {
        void* p = ws + off;
        off = (off + bytes + 255) & ~(size_t)255;
        return p;
    };

    int*      first = (int*)alloc((size_t)kNPTS * 4);
    float*    feat  = (float*)alloc((size_t)kNPTS * 16 * 4);
    unsigned* w1p   = (unsigned*)alloc(16 * 32 * 8 * 4);
    unsigned* w2p   = (unsigned*)alloc(8 * 32 * 8 * 4);
    unsigned* c1p   = (unsigned*)alloc(9 * 8 * 4 * 32 * 8 * 4);
    unsigned* c2p   = (unsigned*)alloc(9 * 8 * 4 * 32 * 8 * 4);
    unsigned* u1p   = (unsigned*)alloc(9 * 4 * 4 * 32 * 8 * 4);
    unsigned* u2p   = (unsigned*)alloc(9 * 1 * 2 * 32 * 8 * 4);
    _Float16* x16   = (_Float16*)alloc((size_t)kTOTAL * 16 * 2);   // 67 MB
    _Float16* a1    = (_Float16*)alloc((size_t)kTOTAL * 16 * 2);   // 67 MB
    _Float16* fused = (_Float16*)alloc((size_t)kTOTAL * 16 * 2);   // 67 MB
    _Float16* ubuf  = a1;  // a1 is dead after conv2; reuse for unet hidden (33 MB)

    init_first<<<(kNPTS + 255) / 256, 256, 0, stream>>>(first);
    seg_min<<<kTOTAL / 256, 256, 0, stream>>>(zbufs, idbufs, isTrain, first);

    pack_mlp_w1<<<16, 256, 0, stream>>>(mlp_w1, w1p);
    pack_mlp_w2<<<8, 256, 0, stream>>>(mlp_w2, w2p);
    pack_conv_w<<<(9 * 8 * 4 * 32 * 8 + 255) / 256, 256, 0, stream>>>(mpn_w1, c1p, 128, 128, 128);
    pack_conv_w<<<(9 * 8 * 4 * 32 * 8 + 255) / 256, 256, 0, stream>>>(mpn_w2, c2p, 128, 128, 128);
    pack_conv_w<<<(9 * 4 * 4 * 32 * 8 + 255) / 256, 256, 0, stream>>>(unet_w1, u1p, 128, 64, 64);
    pack_conv_w<<<(9 * 1 * 2 * 32 * 8 + 255) / 256, 256, 0, stream>>>(unet_w2, u2p, 64, 16, 3);

    mlp_wmma<<<(kNPTS + 63) / 64, 128, 0, stream>>>(zbufs, ray, first, w1p, w2p,
                                                    mlp_b1, mlp_b2, feat);
    gather_feat<<<kTOTAL / 256, 256, 0, stream>>>(zbufs, idbufs, isTrain, feat, (uint4*)x16);

    conv3<128, 128, 0><<<4096, 256, 0, stream>>>(x16,   c1p, mpn_b1,  a1,      nullptr, nullptr);
    conv3<128, 128, 1><<<4096, 256, 0, stream>>>(a1,    c2p, mpn_b2,  fused,   nullptr, x16);
    conv3<128,  64, 0><<<4096, 256, 0, stream>>>(fused, u1p, unet_b1, ubuf,    nullptr, nullptr);
    conv3< 64,  16, 2><<<4096, 256, 0, stream>>>(ubuf,  u2p, unet_b2, nullptr, img,     nullptr);

    (void)in_sizes; (void)n_in; (void)out_size; (void)ws_size;
}
